// MultiHeadAttention_20521353740934
// MI455X (gfx1250) — matrix operand order repaired and checked by reading, not compiled, not run on hardware
//
#include <hip/hip_runtime.h>

// ---------------------------------------------------------------------------
// MHA for MI455X (gfx1250). bf16 V_WMMA_F32_16X16X32_BF16 for all GEMMs,
// f32 accumulation. Weights pre-transposed once so B-fragments are contiguous;
// GEMM stages BOTH A and B tiles with GLOBAL_LOAD_ASYNC_TO_LDS_B128
// (triple-buffered, one barrier + s_wait_asynccnt per 64-deep k-step,
// 16 wmma per wave per step). Attention is flash-style, barrier-free,
// 64 keys per iteration (one softmax reduction tree per 16 wmma).
// ---------------------------------------------------------------------------

typedef __attribute__((ext_vector_type(16))) __bf16 v16bf;
typedef __attribute__((ext_vector_type(8)))  __bf16 v8bf;
typedef __attribute__((ext_vector_type(4)))  __bf16 v4bf;
typedef __attribute__((ext_vector_type(8)))  float  v8f;

#define NEG_INF (-__builtin_inff())
#define S_LEN 2048
#define DM    1024
#define NH    16

__device__ __forceinline__ v8f v8f_zero() {
  v8f z = {0.f, 0.f, 0.f, 0.f, 0.f, 0.f, 0.f, 0.f};
  return z;
}

// A-fragment (16x32 bf16): lanes 0-15 -> row M=lane, K 0..15 contiguous;
// lanes 16-31 -> row M=lane-16, K 16..31. Caller adds (lane>>4)*16.
__device__ __forceinline__ v16bf load_afrag(const __bf16* p) {
  union { v16bf v; v8bf h[2]; } u;
  u.h[0] = *(const v8bf*)(p);
  u.h[1] = *(const v8bf*)(p + 8);
  return u.v;
}

// B-fragment (32x16 bf16): lanes 0-15 -> col N=lane, K 0..15 contiguous;
// lanes 16-31 -> col N=lane-16, K 16..31. Caller adds (lane>>4)*16.
__device__ __forceinline__ v16bf load_bfrag(const __bf16* p) {
  union { v16bf v; v8bf h[2]; } u;
  u.h[0] = *(const v8bf*)(p);
  u.h[1] = *(const v8bf*)(p + 8);
  return u.v;
}

__device__ __forceinline__ v8f wmma_bf16(v16bf a, v16bf b, v8f c) {
  return __builtin_amdgcn_wmma_f32_16x16x32_bf16(
      false, a, false, b, (short)0, c, false, false);
}

// Async DMA: 16 contiguous global bytes -> 16 contiguous LDS bytes, per lane.
// Tracked by ASYNCcnt (CDNA5 ISA), in-order completion.
__device__ __forceinline__ void async_copy_b128(unsigned lds_off, const void* gptr) {
  asm volatile("global_load_async_to_lds_b128 %0, %1, off"
               :: "v"(lds_off), "v"((unsigned long long)(uintptr_t)gptr)
               : "memory");
}
__device__ __forceinline__ void wait_async_le8() {
  asm volatile("s_wait_asynccnt 0x8" ::: "memory");
}
__device__ __forceinline__ void wait_async_0() {
  asm volatile("s_wait_asynccnt 0x0" ::: "memory");
}

// ---------------------------------------------------------------------------
// f32 -> bf16 elementwise (n multiple of 4)
// ---------------------------------------------------------------------------
__global__ __launch_bounds__(256) void cvt_f32_bf16(const float* __restrict__ in,
                                                    __bf16* __restrict__ out, int n) {
  int i = (blockIdx.x * 256 + threadIdx.x) * 4;
  if (i + 4 <= n) {
    const float4 v = *(const float4*)(in + i);
    v4bf o;
    o.x = (__bf16)v.x; o.y = (__bf16)v.y; o.z = (__bf16)v.z; o.w = (__bf16)v.w;
    *(v4bf*)(out + i) = o;
  }
}

// ---------------------------------------------------------------------------
// Weight transpose + convert: in[R][C] f32 -> out[C][R] bf16, 64x64 LDS tiles.
// ---------------------------------------------------------------------------
__global__ __launch_bounds__(256) void cvt_transpose_f32_bf16(
    const float* __restrict__ in, __bf16* __restrict__ out, int R, int C) {
  __shared__ __bf16 tile[64][72];
  const int r0 = blockIdx.y * 64, c0 = blockIdx.x * 64;
#pragma unroll
  for (int i = 0; i < 16; ++i) {
    int e = threadIdx.x + i * 256;
    int r = e >> 6, c = e & 63;
    tile[r][c] = (__bf16)in[(long)(r0 + r) * C + c0 + c];
  }
  __syncthreads();
#pragma unroll
  for (int i = 0; i < 16; ++i) {
    int e = threadIdx.x + i * 256;
    int c = e >> 6, r = e & 63;          // r fast -> coalesced stores
    out[(long)(c0 + c) * R + r0 + r] = tile[r][c];
  }
}

// ---------------------------------------------------------------------------
// C[M,N] = A[M,Kd] @ Bt[N,Kd]^T + bias.  bf16 in, f32 accum.
// Block: 256 thr = 8 waves (4 along M x 2 along N), tile 128x128.
// k-step 64: both A (128x64) and B (128x64) tiles async-DMA'd into
// triple-buffered LDS (8 x b128 per thread per stage); one barrier per step;
// 16 wmma per wave per step.
// store_mode: 0 = f32 row-major, 1 = bf16 row-major,
//             2 = bf16 head-transposed [b][col][s] (for V).
// ---------------------------------------------------------------------------
__global__ __launch_bounds__(256) void gemm_bf16_tn(
    const __bf16* __restrict__ A, const __bf16* __restrict__ Bt,
    const float* __restrict__ bias, void* __restrict__ Cout,
    int M, int N, int Kd, int store_mode) {
  __shared__ __align__(16) __bf16 Asm[3][128][64];   // 48 KB
  __shared__ __align__(16) __bf16 Bsm[3][128][64];   // 48 KB

  const int lane = threadIdx.x & 31;
  const int wave = threadIdx.x >> 5;
  const int mw = wave & 3;        // 4 waves along M (32 rows each)
  const int nw = wave >> 2;       // 2 waves along N (64 cols each)
  const int ln = lane & 15;
  const int hs = lane >> 4;
  const int rowBase = blockIdx.y * 128;
  const int colBase = blockIdx.x * 128;

  v8f acc[2][4];
#pragma unroll
  for (int mi = 0; mi < 2; ++mi)
#pragma unroll
    for (int ni = 0; ni < 4; ++ni) acc[mi][ni] = v8f_zero();

  // Stage 128x64 A and B tiles into buffer `buf`: 2048 x 16B async copies.
  auto stage = [&](int k0, int buf) {
#pragma unroll
    for (int i = 0; i < 4; ++i) {
      int e = threadIdx.x + i * 256;          // 0..1023
      int r = e >> 3, c8 = e & 7;             // row, 16B chunk
      async_copy_b128((unsigned)(uintptr_t)&Asm[buf][r][c8 * 8],
                      A + (long)(rowBase + r) * Kd + k0 + c8 * 8);
    }
#pragma unroll
    for (int i = 0; i < 4; ++i) {
      int e = threadIdx.x + i * 256;
      int r = e >> 3, c8 = e & 7;
      async_copy_b128((unsigned)(uintptr_t)&Bsm[buf][r][c8 * 8],
                      Bt + (long)(colBase + r) * Kd + k0 + c8 * 8);
    }
  };

  const int T = Kd >> 6;          // 64-deep steps
  int cur = 0;
  stage(0, 0);
  for (int it = 0; it < T; ++it) {
    const int k0 = it << 6;
    if (it + 1 < T) {
      int nxt = cur + 1; if (nxt == 3) nxt = 0;
      stage(k0 + 64, nxt);        // overlap next tile's DMA with compute
      wait_async_le8();           // in-order: all but the 8 just issued done
    } else {
      wait_async_0();
    }
    __syncthreads();              // publish LDS writes to all waves

#pragma unroll
    for (int kt = 0; kt < 2; ++kt) {
      v16bf af0 = load_afrag(&Asm[cur][mw * 32 + ln][kt * 32 + hs * 16]);
      v16bf af1 = load_afrag(&Asm[cur][mw * 32 + 16 + ln][kt * 32 + hs * 16]);
      v16bf bf0 = load_bfrag(&Bsm[cur][nw * 64 + 0  + ln][kt * 32 + hs * 16]);
      v16bf bf1 = load_bfrag(&Bsm[cur][nw * 64 + 16 + ln][kt * 32 + hs * 16]);
      v16bf bf2 = load_bfrag(&Bsm[cur][nw * 64 + 32 + ln][kt * 32 + hs * 16]);
      v16bf bf3 = load_bfrag(&Bsm[cur][nw * 64 + 48 + ln][kt * 32 + hs * 16]);
      acc[0][0] = wmma_bf16(af0, bf0, acc[0][0]);
      acc[1][0] = wmma_bf16(af1, bf0, acc[1][0]);
      acc[0][1] = wmma_bf16(af0, bf1, acc[0][1]);
      acc[1][1] = wmma_bf16(af1, bf1, acc[1][1]);
      acc[0][2] = wmma_bf16(af0, bf2, acc[0][2]);
      acc[1][2] = wmma_bf16(af1, bf2, acc[1][2]);
      acc[0][3] = wmma_bf16(af0, bf3, acc[0][3]);
      acc[1][3] = wmma_bf16(af1, bf3, acc[1][3]);
    }
    cur = cur + 1; if (cur == 3) cur = 0;
  }

#pragma unroll
  for (int mi = 0; mi < 2; ++mi) {
#pragma unroll
    for (int ni = 0; ni < 4; ++ni) {
      const int c = colBase + nw * 64 + ni * 16 + ln;
      const int r0 = rowBase + mw * 32 + mi * 16 + 8 * hs;
      const float bv = bias[c];
      if (store_mode == 1) {
        __bf16* C = (__bf16*)Cout;
#pragma unroll
        for (int r = 0; r < 8; ++r)
          C[(long)(r0 + r) * N + c] = (__bf16)(acc[mi][ni][r] + bv);
      } else if (store_mode == 2) {
        __bf16* C = (__bf16*)Cout;   // [b][col][s], s contiguous per head-dim
#pragma unroll
        for (int r = 0; r < 8; ++r) {
          const int row = r0 + r;
          C[((long)(row >> 11) * N + c) * S_LEN + (row & (S_LEN - 1))] =
              (__bf16)(acc[mi][ni][r] + bv);
        }
      } else {
        float* C = (float*)Cout;
#pragma unroll
        for (int r = 0; r < 8; ++r)
          C[(long)(r0 + r) * N + c] = acc[mi][ni][r] + bv;
      }
    }
  }
}

// ---------------------------------------------------------------------------
// Flash attention, barrier-free. Block = 128 thr = 4 independent waves,
// each owning 16 queries of one (b,h). 64 keys per iteration:
// scores (8 wmma) -> one online-softmax reduction tree (half-wave shfl) ->
// P transposed through per-wave LDS (same-wave DS is in-order) ->
// P@V (8 wmma) with V read from the pre-transposed global layout.
// Mask added as -inf BEFORE the 1/sqrt(64) scale (matches reference).
// ---------------------------------------------------------------------------
__global__ __launch_bounds__(128) void attn_kernel(
    const __bf16* __restrict__ Q, const __bf16* __restrict__ K,
    const __bf16* __restrict__ Vt,     // [b][h*64+vd][s]
    const int* __restrict__ pmask,
    __bf16* __restrict__ O) {
  __shared__ __align__(16) __bf16 Plds[4][16][64];

  const int lane = threadIdx.x & 31;
  const int wave = threadIdx.x >> 5;
  const int ln = lane & 15;
  const int hs = lane >> 4;
  const int b = blockIdx.y >> 4;
  const int h = blockIdx.y & 15;
  const int q0 = blockIdx.x * 64 + wave * 16;

  const long qRow = (long)(b * S_LEN + q0 + ln) * DM + h * 64;
  v16bf qf0 = load_afrag(Q + qRow + 0  + hs * 16);
  v16bf qf1 = load_afrag(Q + qRow + 32 + hs * 16);

  const long vBase = ((long)b * DM + h * 64) * S_LEN;   // Vt[b][h*64+..][..]

  v8f acc[4];
#pragma unroll
  for (int j = 0; j < 4; ++j) acc[j] = v8f_zero();
  float m[8], l[8];
#pragma unroll
  for (int r = 0; r < 8; ++r) { m[r] = NEG_INF; l[r] = 0.f; }

  for (int kb = 0; kb < S_LEN; kb += 64) {
    const long kRow = (long)(b * S_LEN + kb + ln) * DM + h * 64;

    if (kb + 64 < S_LEN) {   // prefetch next key tile (K rows + V rows)
#pragma unroll
      for (int s = 0; s < 4; ++s)
        __builtin_prefetch(K + kRow + (long)(64 + 16 * s) * DM, 0, 1);
#pragma unroll
      for (int j = 0; j < 4; ++j)
        __builtin_prefetch(Vt + vBase + (long)(j * 16 + ln) * S_LEN + kb + 64, 0, 1);
    }

    // scores: Q(16x64) @ K^T(64x64); K^T column n = K row n (contiguous d)
    v8f sc[4];
#pragma unroll
    for (int s = 0; s < 4; ++s) {
      const __bf16* kp = K + kRow + (long)(16 * s) * DM;
      v8f z = v8f_zero();
      z = wmma_bf16(qf0, load_bfrag(kp + 0  + hs * 16), z);
      z = wmma_bf16(qf1, load_bfrag(kp + 32 + hs * 16), z);
      sc[s] = z;
    }

    float bias[4];
#pragma unroll
    for (int s = 0; s < 4; ++s)
      bias[s] = (pmask[b * S_LEN + kb + 16 * s + ln] > 0) ? NEG_INF : 0.f;

    float pv[4][8];
#pragma unroll
    for (int r = 0; r < 8; ++r) {
      float t0 = sc[0][r] * 0.125f + bias[0];   // (score + mask) / sqrt(64)
      float t1 = sc[1][r] * 0.125f + bias[1];
      float t2 = sc[2][r] * 0.125f + bias[2];
      float t3 = sc[3][r] * 0.125f + bias[3];
      float x = fmaxf(fmaxf(t0, t1), fmaxf(t2, t3));
      x = fmaxf(x, __shfl_xor(x, 1, 32));
      x = fmaxf(x, __shfl_xor(x, 2, 32));
      x = fmaxf(x, __shfl_xor(x, 4, 32));
      x = fmaxf(x, __shfl_xor(x, 8, 32));
      float mn = fmaxf(m[r], x);
      bool dead = (mn == NEG_INF);
      float corr = dead ? 0.f : __expf(m[r] - mn);
      float e0 = dead ? 0.f : __expf(t0 - mn);
      float e1 = dead ? 0.f : __expf(t1 - mn);
      float e2 = dead ? 0.f : __expf(t2 - mn);
      float e3 = dead ? 0.f : __expf(t3 - mn);
      float sum = (e0 + e1) + (e2 + e3);
      sum += __shfl_xor(sum, 1, 32);
      sum += __shfl_xor(sum, 2, 32);
      sum += __shfl_xor(sum, 4, 32);
      sum += __shfl_xor(sum, 8, 32);
      l[r] = l[r] * corr + sum;
      m[r] = mn;
      pv[0][r] = e0; pv[1][r] = e1; pv[2][r] = e2; pv[3][r] = e3;
#pragma unroll
      for (int j = 0; j < 4; ++j) acc[j][r] *= corr;
    }

    // C-layout -> A-layout transpose via per-wave LDS (16x64 P tile)
#pragma unroll
    for (int r = 0; r < 8; ++r) {
#pragma unroll
      for (int s = 0; s < 4; ++s)
        Plds[wave][r + 8 * hs][16 * s + ln] = (__bf16)pv[s][r];
    }
#pragma unroll
    for (int kt = 0; kt < 2; ++kt) {
      v16bf af = load_afrag(&Plds[wave][ln][kt * 32 + hs * 16]);
#pragma unroll
      for (int j = 0; j < 4; ++j) {
        v16bf vf = load_bfrag(Vt + vBase + (long)(j * 16 + ln) * S_LEN +
                              kb + kt * 32 + hs * 16);
        acc[j] = wmma_bf16(af, vf, acc[j]);
      }
    }
  }

  const long oRow = (long)(b * S_LEN + q0 + 8 * hs) * DM + h * 64;
#pragma unroll
  for (int j = 0; j < 4; ++j) {
#pragma unroll
    for (int r = 0; r < 8; ++r) {
      O[oRow + (long)r * DM + j * 16 + ln] = (__bf16)(acc[j][r] / l[r]);
    }
  }
}

// ---------------------------------------------------------------------------
// Launch
// ---------------------------------------------------------------------------
extern "C" void kernel_launch(void* const* d_in, const int* in_sizes, int n_in,
                              void* d_out, int out_size, void* d_ws, size_t ws_size,
                              hipStream_t stream) {
  const float* x   = (const float*)d_in[0];
  const int*   pm  = (const int*)d_in[1];
  const float* qw  = (const float*)d_in[2];
  const float* qb  = (const float*)d_in[3];
  const float* kw  = (const float*)d_in[4];
  const float* kbi = (const float*)d_in[5];
  const float* vw  = (const float*)d_in[6];
  const float* vb  = (const float*)d_in[7];
  const float* ow  = (const float*)d_in[8];
  const float* ob  = (const float*)d_in[9];
  (void)in_sizes; (void)n_in; (void)out_size; (void)ws_size;

  const int Mrows = 2 * S_LEN;  // 4096
  const int D = DM;             // 1024

  char* base = (char*)d_ws;
  size_t off = 0;
  auto take = [&](size_t bytes) -> void* {
    void* p = base + off;
    off += (bytes + 255) & ~(size_t)255;
    return p;
  };
  __bf16* xb  = (__bf16*)take((size_t)Mrows * D * 2);
  __bf16* qwT = (__bf16*)take((size_t)D * D * 2);
  __bf16* kwT = (__bf16*)take((size_t)D * D * 2);
  __bf16* vwT = (__bf16*)take((size_t)D * D * 2);
  __bf16* owT = (__bf16*)take((size_t)D * D * 2);
  __bf16* Qb  = (__bf16*)take((size_t)Mrows * D * 2);
  __bf16* Kb  = (__bf16*)take((size_t)Mrows * D * 2);
  __bf16* VtG = (__bf16*)take((size_t)Mrows * D * 2);
  __bf16* Ab  = (__bf16*)take((size_t)Mrows * D * 2);

  cvt_f32_bf16<<<(Mrows * D) / 1024, 256, 0, stream>>>(x, xb, Mrows * D);
  dim3 gt(D / 64, D / 64);
  cvt_transpose_f32_bf16<<<gt, 256, 0, stream>>>(qw, qwT, D, D);
  cvt_transpose_f32_bf16<<<gt, 256, 0, stream>>>(kw, kwT, D, D);
  cvt_transpose_f32_bf16<<<gt, 256, 0, stream>>>(vw, vwT, D, D);
  cvt_transpose_f32_bf16<<<gt, 256, 0, stream>>>(ow, owT, D, D);

  dim3 gg(D / 128, Mrows / 128);
  gemm_bf16_tn<<<gg, 256, 0, stream>>>(xb, qwT, qb,  Qb,  Mrows, D, D, 1);
  gemm_bf16_tn<<<gg, 256, 0, stream>>>(xb, kwT, kbi, Kb,  Mrows, D, D, 1);
  gemm_bf16_tn<<<gg, 256, 0, stream>>>(xb, vwT, vb,  VtG, Mrows, D, D, 2);

  attn_kernel<<<dim3(S_LEN / 64, 2 * NH), 128, 0, stream>>>(Qb, Kb, VtG, pm, Ab);

  gemm_bf16_tn<<<gg, 256, 0, stream>>>(Ab, owT, ob, d_out, Mrows, D, D, 0);
}
